// CausalSelfAttention_12352325943694
// MI455X (gfx1250) — matrix operand order repaired and checked by reading, not compiled, not run on hardware
//
#include <hip/hip_runtime.h>
#include <hip/hip_bf16.h>
#include <stdint.h>

#define BB 2
#define TT 2048
#define DD 1024
#define HH 16
#define HD 64

typedef __attribute__((ext_vector_type(16))) __bf16 v16bf;
typedef __attribute__((ext_vector_type(8)))  float  v8f;

union Frag {
    uint4 u[2];
    v16bf v;
};

__device__ __forceinline__ unsigned short f2bf(float f) {
    union { float f; unsigned int u; } c;
    c.f = f;
    unsigned int r = 0x7FFFu + ((c.u >> 16) & 1u);
    return (unsigned short)((c.u + r) >> 16);
}

// ---------------- fp32 -> bf16 conversion ----------------
__global__ void cvt_bf16_kernel(const float* __restrict__ src,
                                unsigned short* __restrict__ dst, int n) {
    int i = (blockIdx.x * blockDim.x + threadIdx.x) * 4;
    if (i >= n) return;
    float4 f = *(const float4*)(src + i);
    union { unsigned short s[4]; uint2 u; } p;
    p.s[0] = f2bf(f.x); p.s[1] = f2bf(f.y);
    p.s[2] = f2bf(f.z); p.s[3] = f2bf(f.w);
    *(uint2*)(dst + i) = p.u;
}

// ---------------- W-tile staging: [32 k][64 n] -> LDS transposed [n][k] ----------------
__device__ __forceinline__ void stage_w(const unsigned short* __restrict__ W,
                                        int N, int n0, int kk,
                                        unsigned short* __restrict__ buf /* [64][40] */) {
    int tid = threadIdx.x;
    int kl = tid >> 3;            // 0..31
    int n8 = (tid & 7) * 8;       // 0..56
    union { uint4 u; unsigned short s[8]; } wv;
    wv.u = *(const uint4*)(W + (size_t)(kk + kl) * N + n0 + n8);
    #pragma unroll
    for (int i = 0; i < 8; ++i) buf[(n8 + i) * 40 + kl] = wv.s[i];
}

// ---------------- shared GEMM mainloop: 256x64 per block, 32x64 per wave ----------------
// A: [M, DD] bf16 row-major.  W: [DD, N] bf16 row-major.  Double-buffered W in LDS.
__device__ __forceinline__ void gemm_mainloop(const unsigned short* __restrict__ A,
                                              const unsigned short* __restrict__ W,
                                              int N, int m0, int n0,
                                              unsigned short* __restrict__ ldsW /* [2][64*40] */,
                                              v8f (&acc)[2][4]) {
    const int lane = threadIdx.x & 31;
    const int wave = threadIdx.x >> 5;
    const int hl   = lane >> 4;
    const int ln   = lane & 15;
    const unsigned short* arow0 = A + (size_t)(m0 + wave * 32 + ln) * DD;
    const unsigned short* arow1 = arow0 + (size_t)16 * DD;

    stage_w(W, N, n0, 0, ldsW);

    for (int kk = 0; kk < DD; kk += 32) {
        __syncthreads();
        unsigned short* cur = ldsW + ((kk >> 5) & 1) * (64 * 40);
        if (kk + 32 < DD) {
            stage_w(W, N, n0, kk + 32, ldsW + (((kk >> 5) & 1) ^ 1) * (64 * 40));
            __builtin_prefetch(arow0 + kk + 32, 0, 1);
            __builtin_prefetch(arow1 + kk + 32, 0, 1);
        }

        // A fragments: two 16x32 row tiles
        Frag a0, a1;
        a0.u[0] = *(const uint4*)(arow0 + kk + hl * 8);
        a0.u[1] = *(const uint4*)(arow0 + kk + 16 + hl * 8);
        a1.u[0] = *(const uint4*)(arow1 + kk + hl * 8);
        a1.u[1] = *(const uint4*)(arow1 + kk + 16 + hl * 8);

        // batch all 4 B fragments into distinct regs, then burst 8 WMMAs
        Frag b[4];
        #pragma unroll
        for (int nt = 0; nt < 4; ++nt) {
            const uint4* bp = (const uint4*)&cur[(nt * 16 + ln) * 40 + hl * 8];
            b[nt].u[0] = bp[0]; b[nt].u[1] = bp[2];
        }
        #pragma unroll
        for (int nt = 0; nt < 4; ++nt) {
            acc[0][nt] = __builtin_amdgcn_wmma_f32_16x16x32_bf16(
                false, a0.v, false, b[nt].v, (short)0, acc[0][nt], false, false);
            acc[1][nt] = __builtin_amdgcn_wmma_f32_16x16x32_bf16(
                false, a1.v, false, b[nt].v, (short)0, acc[1][nt], false, false);
        }
    }
}

// ---------------- QKV GEMM: out scattered to q/k/v [B,H,T,HD] bf16 ----------------
__global__ __launch_bounds__(256)
void qkv_gemm_kernel(const unsigned short* __restrict__ xb,
                     const unsigned short* __restrict__ wb,
                     const float* __restrict__ bias,
                     unsigned short* __restrict__ qb,
                     unsigned short* __restrict__ kb,
                     unsigned short* __restrict__ vb) {
    __shared__ unsigned short ldsW[2 * 64 * 40];
    const int nblks = (3 * DD) / 64;           // 48
    const int m0 = (blockIdx.x / nblks) * 256;
    const int n0 = (blockIdx.x % nblks) * 64;

    v8f acc[2][4];
    #pragma unroll
    for (int rt = 0; rt < 2; ++rt)
        #pragma unroll
        for (int t = 0; t < 4; ++t)
            #pragma unroll
            for (int r = 0; r < 8; ++r) acc[rt][t][r] = 0.f;

    gemm_mainloop(xb, wb, 3 * DD, m0, n0, ldsW, acc);

    const int lane = threadIdx.x & 31, wave = threadIdx.x >> 5;
    const int hl = lane >> 4, ln = lane & 15;
    #pragma unroll
    for (int nt = 0; nt < 4; ++nt) {
        int col   = n0 + nt * 16 + ln;
        int which = col >> 10;                 // uniform across tile
        int dp    = col & 1023;
        int h = dp >> 6, hd = dp & 63;
        unsigned short* dst = (which == 0) ? qb : (which == 1) ? kb : vb;
        float bv = bias[col];
        #pragma unroll
        for (int rt = 0; rt < 2; ++rt)
            #pragma unroll
            for (int r = 0; r < 8; ++r) {
                int m = m0 + wave * 32 + rt * 16 + r + 8 * hl;
                int b = m >> 11, t = m & (TT - 1);
                dst[((size_t)(b * HH + h) * TT + t) * HD + hd] = f2bf(acc[rt][nt][r] + bv);
            }
    }
}

// ---------------- Flash attention: one block = (b,h) x 128 q-rows ----------------
__device__ __forceinline__ void stage_kv(const unsigned short* __restrict__ khead,
                                         const unsigned short* __restrict__ vhead,
                                         int kblk,
                                         unsigned short* __restrict__ bK /* [32][80] */,
                                         unsigned short* __restrict__ bVT /* [64][40] */) {
    int tid = threadIdx.x;
    int key = tid >> 3;                     // 0..31
    int d0  = (tid & 7) * 8;                // 0..56
    size_t grow = (size_t)(kblk * 32 + key) * HD + d0;
    *(uint4*)&bK[key * 80 + d0] = *(const uint4*)(khead + grow);
    union { uint4 u; unsigned short s[8]; } vv;
    vv.u = *(const uint4*)(vhead + grow);
    #pragma unroll
    for (int i = 0; i < 8; ++i) bVT[(d0 + i) * 40 + key] = vv.s[i];
}

__global__ __launch_bounds__(256)
void attn_kernel(const unsigned short* __restrict__ qbuf,
                 const unsigned short* __restrict__ kbuf,
                 const unsigned short* __restrict__ vbuf,
                 unsigned short* __restrict__ attb) {
    __shared__ unsigned short ldsK[2 * 32 * 80];    // [buf][key][d]
    __shared__ unsigned short ldsVT[2 * 64 * 40];   // [buf][d][key]
    __shared__ unsigned short ldsP[8 * 16 * 40];    // per-wave [row][key]

    const int nqb = TT / 128;                       // 16
    const int qb  = blockIdx.x % nqb;
    const int bh  = blockIdx.x / nqb;               // b*H + h
    const int lane = threadIdx.x & 31, wave = threadIdx.x >> 5;
    const int hl = lane >> 4, ln = lane & 15;
    const int q0 = qb * 128 + wave * 16;

    const unsigned short* qhead = qbuf + (size_t)bh * TT * HD;
    const unsigned short* khead = kbuf + (size_t)bh * TT * HD;
    const unsigned short* vhead = vbuf + (size_t)bh * TT * HD;

    Frag qa0, qa1;
    {
        const unsigned short* qr = qhead + (size_t)(q0 + ln) * HD;
        qa0.u[0] = *(const uint4*)(qr + hl * 8);
        qa0.u[1] = *(const uint4*)(qr + 16 + hl * 8);
        qa1.u[0] = *(const uint4*)(qr + 32 + hl * 8);
        qa1.u[1] = *(const uint4*)(qr + 48 + hl * 8);
    }

    v8f acc[4];
    float mrow[8], lrow[8];
    #pragma unroll
    for (int r = 0; r < 8; ++r) { mrow[r] = -1e30f; lrow[r] = 0.f; }
    #pragma unroll
    for (int t = 0; t < 4; ++t)
        #pragma unroll
        for (int r = 0; r < 8; ++r) acc[t][r] = 0.f;

    const int kbcount = (qb + 1) * 4;               // causal bound (32-key blocks)
    stage_kv(khead, vhead, 0, ldsK, ldsVT);

    for (int kblk = 0; kblk < kbcount; ++kblk) {
        __syncthreads();
        const int cur = kblk & 1;
        unsigned short* cK  = ldsK  + cur * (32 * 80);
        unsigned short* cVT = ldsVT + cur * (64 * 40);
        if (kblk + 1 < kbcount)
            stage_kv(khead, vhead, kblk + 1,
                     ldsK + (cur ^ 1) * (32 * 80), ldsVT + (cur ^ 1) * (64 * 40));

        if (kblk * 32 <= q0 + 15) {                 // wave-uniform: EXEC stays full
            // ---- S = Q @ K^T: batch 4 B frags, then 4 WMMAs ----
            Frag sb[4];
            #pragma unroll
            for (int j = 0; j < 2; ++j)
                #pragma unroll
                for (int dh = 0; dh < 2; ++dh) {
                    const uint4* bp = (const uint4*)&cK[(j * 16 + ln) * 80 + dh * 32 + hl * 8];
                    sb[j * 2 + dh].u[0] = bp[0];
                    sb[j * 2 + dh].u[1] = bp[2];
                }
            v8f S[2];
            #pragma unroll
            for (int j = 0; j < 2; ++j) {
                v8f s;
                #pragma unroll
                for (int r = 0; r < 8; ++r) s[r] = 0.f;
                s = __builtin_amdgcn_wmma_f32_16x16x32_bf16(
                    false, qa0.v, false, sb[j * 2].v, (short)0, s, false, false);
                s = __builtin_amdgcn_wmma_f32_16x16x32_bf16(
                    false, qa1.v, false, sb[j * 2 + 1].v, (short)0, s, false, false);
                S[j] = s;
            }
            // ---- causal mask + scale ----
            const float sc = 0.125f;                // 1/sqrt(64)
            #pragma unroll
            for (int j = 0; j < 2; ++j) {
                int col = kblk * 32 + j * 16 + ln;
                #pragma unroll
                for (int r = 0; r < 8; ++r) {
                    int row = q0 + r + 8 * hl;
                    float v = S[j][r] * sc;
                    S[j][r] = (col <= row) ? v : -1e30f;
                }
            }
            // ---- online softmax (row spread across 16 lanes of a half-wave) ----
            #pragma unroll
            for (int r = 0; r < 8; ++r) {
                float mt = fmaxf(S[0][r], S[1][r]);
                for (int msk = 1; msk < 16; msk <<= 1)
                    mt = fmaxf(mt, __shfl_xor(mt, msk, 32));
                float mn = fmaxf(mrow[r], mt);
                float scale = __expf(mrow[r] - mn);
                mrow[r] = mn;
                float p0 = __expf(S[0][r] - mn);
                float p1 = __expf(S[1][r] - mn);
                S[0][r] = p0; S[1][r] = p1;
                float rs = p0 + p1;
                for (int msk = 1; msk < 16; msk <<= 1)
                    rs += __shfl_xor(rs, msk, 32);
                lrow[r] = lrow[r] * scale + rs;
                #pragma unroll
                for (int t = 0; t < 4; ++t) acc[t][r] *= scale;
            }
            // ---- bounce P through per-wave LDS: C layout -> A layout ----
            unsigned short* pw = &ldsP[wave * 16 * 40];
            #pragma unroll
            for (int j = 0; j < 2; ++j)
                #pragma unroll
                for (int r = 0; r < 8; ++r)
                    pw[(r + 8 * hl) * 40 + j * 16 + ln] = f2bf(S[j][r]);
            Frag pa;   // LDS in-order within wave: RAW safe
            pa.u[0] = *(const uint4*)&pw[ln * 40 + hl * 8];
            pa.u[1] = *(const uint4*)&pw[ln * 40 + 16 + hl * 8];
            // ---- acc += P @ V: batch 4 B frags, then 4 WMMAs ----
            Frag vb[4];
            #pragma unroll
            for (int nt = 0; nt < 4; ++nt) {
                const uint4* bp = (const uint4*)&cVT[(nt * 16 + ln) * 40 + hl * 8];
                vb[nt].u[0] = bp[0]; vb[nt].u[1] = bp[2];
            }
            #pragma unroll
            for (int nt = 0; nt < 4; ++nt)
                acc[nt] = __builtin_amdgcn_wmma_f32_16x16x32_bf16(
                    false, pa.v, false, vb[nt].v, (short)0, acc[nt], false, false);
        }
    }

    // ---- normalize + write [b,t,h*64+d] bf16 ----
    const int b = bh / HH, h = bh % HH;
    #pragma unroll
    for (int r = 0; r < 8; ++r) {
        float inv = 1.0f / lrow[r];
        int trow = q0 + r + 8 * hl;
        size_t base = ((size_t)(b * TT + trow)) * DD + h * HD;
        #pragma unroll
        for (int nt = 0; nt < 4; ++nt)
            attb[base + nt * 16 + ln] = f2bf(acc[nt][r] * inv);
    }
}

// ---------------- Projection GEMM: fp32 out + bias ----------------
__global__ __launch_bounds__(256)
void proj_gemm_kernel(const unsigned short* __restrict__ attb,
                      const unsigned short* __restrict__ wb,
                      const float* __restrict__ bias,
                      float* __restrict__ out) {
    __shared__ unsigned short ldsW[2 * 64 * 40];
    const int nblks = DD / 64;                 // 16
    const int m0 = (blockIdx.x / nblks) * 256;
    const int n0 = (blockIdx.x % nblks) * 64;

    v8f acc[2][4];
    #pragma unroll
    for (int rt = 0; rt < 2; ++rt)
        #pragma unroll
        for (int t = 0; t < 4; ++t)
            #pragma unroll
            for (int r = 0; r < 8; ++r) acc[rt][t][r] = 0.f;

    gemm_mainloop(attb, wb, DD, m0, n0, ldsW, acc);

    const int lane = threadIdx.x & 31, wave = threadIdx.x >> 5;
    const int hl = lane >> 4, ln = lane & 15;
    #pragma unroll
    for (int nt = 0; nt < 4; ++nt) {
        int col = n0 + nt * 16 + ln;
        float bv = bias[col];
        #pragma unroll
        for (int rt = 0; rt < 2; ++rt)
            #pragma unroll
            for (int r = 0; r < 8; ++r) {
                int m = m0 + wave * 32 + rt * 16 + r + 8 * hl;
                out[(size_t)m * DD + col] = acc[rt][nt][r] + bv;
            }
    }
}

// ---------------- host ----------------
extern "C" void kernel_launch(void* const* d_in, const int* in_sizes, int n_in,
                              void* d_out, int out_size, void* d_ws, size_t ws_size,
                              hipStream_t stream) {
    const float* x      = (const float*)d_in[0];
    const float* w_qkv  = (const float*)d_in[1];
    const float* b_qkv  = (const float*)d_in[2];
    const float* w_proj = (const float*)d_in[3];
    const float* b_proj = (const float*)d_in[4];
    float* out = (float*)d_out;

    char* ws = (char*)d_ws;
    unsigned short* xb     = (unsigned short*)ws; ws += (size_t)BB * TT * DD * 2;
    unsigned short* wqkvb  = (unsigned short*)ws; ws += (size_t)DD * 3 * DD * 2;
    unsigned short* wprojb = (unsigned short*)ws; ws += (size_t)DD * DD * 2;
    unsigned short* qbuf   = (unsigned short*)ws; ws += (size_t)BB * HH * TT * HD * 2;
    unsigned short* kbuf   = (unsigned short*)ws; ws += (size_t)BB * HH * TT * HD * 2;
    unsigned short* vbuf   = (unsigned short*)ws; ws += (size_t)BB * HH * TT * HD * 2;
    unsigned short* attb   = (unsigned short*)ws; ws += (size_t)BB * TT * DD * 2;

    int n;
    n = BB * TT * DD;
    cvt_bf16_kernel<<<n / 1024, 256, 0, stream>>>(x, xb, n);
    n = DD * 3 * DD;
    cvt_bf16_kernel<<<n / 1024, 256, 0, stream>>>(w_qkv, wqkvb, n);
    n = DD * DD;
    cvt_bf16_kernel<<<n / 1024, 256, 0, stream>>>(w_proj, wprojb, n);

    qkv_gemm_kernel<<<(BB * TT / 256) * (3 * DD / 64), 256, 0, stream>>>(
        xb, wqkvb, b_qkv, qbuf, kbuf, vbuf);

    attn_kernel<<<BB * HH * (TT / 128), 256, 0, stream>>>(qbuf, kbuf, vbuf, attb);

    proj_gemm_kernel<<<(BB * TT / 256) * (DD / 64), 256, 0, stream>>>(
        attb, wprojb, b_proj, out);
}
